// AuxLossFreeMoE_52939766890807
// MI455X (gfx1250) — matrix operand order repaired and checked by reading, not compiled, not run on hardware
//
#include <hip/hip_runtime.h>
#include <hip/hip_bf16.h>

// ---------------- problem constants ----------------
static constexpr int Ttok = 4096;   // B*S
static constexpr int HD   = 2048;   // hidden
static constexpr int NE   = 16;     // experts
static constexpr int TOPK = 2;
static constexpr int ID   = 1024;   // routed intermediate
static constexpr int ISD  = 2048;   // shared intermediate
static constexpr float RATIO = 0.1f;
static constexpr float EPS   = 1e-9f;

// ---------------- WMMA types ----------------
typedef __attribute__((ext_vector_type(16))) __bf16 v16bf;
typedef __attribute__((ext_vector_type(8)))  float  v8f;

union FragBF { v16bf v; uint4 q[2]; };

__device__ __forceinline__ v8f wmma_bf16(v16bf a, v16bf b, v8f c) {
  // D(f32 16x16) = A(bf16 16x32) x B(bf16 32x16) + C
  return __builtin_amdgcn_wmma_f32_16x16x32_bf16(false, a, false, b, (short)0, c, false, false);
}

// A fragment: lane holds row m=(lane&15); K-chunks [kk+kb..kk+kb+7] and [kk+16+kb..+7], kb=(lane>>4)*8
__device__ __forceinline__ v16bf load_a_frag(const __bf16* rowPtr, int kk, int lane) {
  FragBF f;
  const int kb = (lane >> 4) * 8;
  f.q[0] = *reinterpret_cast<const uint4*>(rowPtr + kk + kb);
  f.q[1] = *reinterpret_cast<const uint4*>(rowPtr + kk + 16 + kb);
  return f.v;
}

// B fragment from W^T stored [N][Kd]: lane col n=nBase+(lane&15); K-chunks [kk+kb..kk+kb+7] and [kk+16+kb..+7], kb=(lane>>4)*8
__device__ __forceinline__ v16bf load_b_frag(const __bf16* wt, int ldk, int nBase, int kk, int lane) {
  const __bf16* p = wt + (size_t)(nBase + (lane & 15)) * ldk + kk + ((lane >> 4) << 3);
  FragBF f;
  f.q[0] = reinterpret_cast<const uint4*>(p)[0];
  f.q[1] = reinterpret_cast<const uint4*>(p)[2];
  return f.v;
}

__device__ __forceinline__ float silu_f(float x) { return x / (1.0f + __expf(-x)); }

// ---------------- kernel 0: zero expert counters ----------------
__global__ void k_zero_counts(int* counts) {
  if (threadIdx.x < NE) counts[threadIdx.x] = 0;
}

// ---------------- kernel 1: x fp32 -> bf16 ----------------
__global__ void k_cvt_x(const float* __restrict__ x, __bf16* __restrict__ xb, int n) {
  int i = blockIdx.x * 256 + threadIdx.x;
  if (i < n) xb[i] = (__bf16)x[i];
}

// ---------------- kernel 2: tiled transpose + convert: in[R][C] f32 -> out[C][R] bf16 ----------------
__global__ void __launch_bounds__(256) k_transpose_cvt(const float* __restrict__ in,
                                                       __bf16* __restrict__ out,
                                                       int R, int C) {
  __shared__ float tile[32][33];
  const size_t batchOff = (size_t)blockIdx.z * (size_t)R * (size_t)C;
  const float* ib = in + batchOff;
  __bf16* ob = out + batchOff;
  const int x = threadIdx.x, y = threadIdx.y;
  const int bx = blockIdx.x * 32, by = blockIdx.y * 32;
#pragma unroll
  for (int j = 0; j < 32; j += 8)
    tile[y + j][x] = ib[(size_t)(by + y + j) * C + bx + x];
  __syncthreads();
#pragma unroll
  for (int j = 0; j < 32; j += 8)
    ob[(size_t)(bx + y + j) * R + by + x] = (__bf16)tile[x][y + j];
}

// ---------------- kernel 3: routing (sigmoid gate, biased top-2, clean normalized weights) ----------------
__global__ void __launch_bounds__(128) k_routing(const float* __restrict__ x,
                                                 const float* __restrict__ cent,
                                                 const float* __restrict__ bias,
                                                 float* __restrict__ wsel,
                                                 int* __restrict__ counts,
                                                 int* __restrict__ lists) {
  __shared__ float part[128];
  __shared__ float aff[NE];
  const int t = blockIdx.x;
  const int tid = threadIdx.x;
  const int e = tid >> 3, r = tid & 7;
  const float* xr = x + (size_t)t * HD;
  const float* cr = cent + (size_t)e * HD;
  float s = 0.0f;
  for (int c = r; c < HD; c += 8) s += xr[c] * cr[c];
  part[tid] = s;
  __syncthreads();
  if (r == 0) {
    float a = 0.0f;
#pragma unroll
    for (int i = 0; i < 8; ++i) a += part[e * 8 + i];
    aff[e] = 1.0f / (1.0f + __expf(-a));
  }
  __syncthreads();
  if (tid == 0) {
    int i0 = 0, i1 = -1;
    float b0 = -1e30f, b1 = -1e30f;
#pragma unroll
    for (int i = 0; i < NE; ++i) {
      float b = aff[i] + bias[i];
      if (b > b0) { b1 = b0; i1 = i0; b0 = b; i0 = i; }
      else if (b > b1) { b1 = b; i1 = i; }
    }
    const float a0 = aff[i0], a1 = aff[i1];
    const float inv = 1.0f / (a0 + a1 + EPS);
    wsel[2 * t + 0] = a0 * inv;
    wsel[2 * t + 1] = a1 * inv;
    int p0 = atomicAdd(&counts[i0], 1);
    lists[i0 * Ttok + p0] = 2 * t + 0;
    int p1 = atomicAdd(&counts[i1], 1);
    lists[i1 * Ttok + p1] = 2 * t + 1;
  }
}

// ---------------- kernel 4: shared-expert fused gate+up GEMM + SwiGLU epilogue ----------------
// Wave tile: 32M x 64N, dual outputs (gate & up). hs[T][ISD] = silu(xb@wg_s)*(xb@wu_s)
__global__ void __launch_bounds__(128) k_gateup_shared(const __bf16* __restrict__ xb,
                                                       const __bf16* __restrict__ wgT,  // [ISD][HD]
                                                       const __bf16* __restrict__ wuT,
                                                       __bf16* __restrict__ hs) {
  const int lane = threadIdx.x & 31, wave = threadIdx.x >> 5;
  const int mBase = blockIdx.y * 32;
  const int nBase = blockIdx.x * 256 + wave * 64;
  const __bf16* aRow0 = xb + (size_t)(mBase + (lane & 15)) * HD;
  const __bf16* aRow1 = aRow0 + (size_t)16 * HD;
  const v8f zero = {0.f,0.f,0.f,0.f,0.f,0.f,0.f,0.f};
  v8f accg[2][4], accu[2][4];
#pragma unroll
  for (int s = 0; s < 2; ++s)
#pragma unroll
    for (int j = 0; j < 4; ++j) { accg[s][j] = zero; accu[s][j] = zero; }
  for (int kk = 0; kk < HD; kk += 32) {
    __builtin_prefetch(aRow0 + kk + 64, 0, 1);   // global_prefetch_b8
    v16bf a0 = load_a_frag(aRow0, kk, lane);
    v16bf a1 = load_a_frag(aRow1, kk, lane);
#pragma unroll
    for (int j = 0; j < 4; ++j) {
      v16bf bg = load_b_frag(wgT, HD, nBase + j * 16, kk, lane);
      accg[0][j] = wmma_bf16(a0, bg, accg[0][j]);
      accg[1][j] = wmma_bf16(a1, bg, accg[1][j]);
      v16bf bu = load_b_frag(wuT, HD, nBase + j * 16, kk, lane);
      accu[0][j] = wmma_bf16(a0, bu, accu[0][j]);
      accu[1][j] = wmma_bf16(a1, bu, accu[1][j]);
    }
  }
  const int mOff = (lane >> 4) * 8;
  const int nCol = lane & 15;
#pragma unroll
  for (int s = 0; s < 2; ++s)
#pragma unroll
    for (int j = 0; j < 4; ++j)
#pragma unroll
      for (int r = 0; r < 8; ++r) {
        const int m = mBase + s * 16 + mOff + r;
        const int n = nBase + j * 16 + nCol;
        hs[(size_t)m * ISD + n] = (__bf16)(silu_f(accg[s][j][r]) * accu[s][j][r]);
      }
}

// ---------------- kernel 5: shared-expert down-proj -> d_out (scaled by RATIO) ----------------
// Wave tile: 64M x 64N (each B fragment feeds 4 WMMAs).
__global__ void __launch_bounds__(128) k_down_shared(const __bf16* __restrict__ hs,
                                                     const __bf16* __restrict__ wdT,  // [HD][ISD]
                                                     float* __restrict__ out) {
  const int lane = threadIdx.x & 31, wave = threadIdx.x >> 5;
  const int mBase = blockIdx.y * 64;
  const int nBase = blockIdx.x * 256 + wave * 64;
  const __bf16* aRow[4];
#pragma unroll
  for (int s = 0; s < 4; ++s)
    aRow[s] = hs + (size_t)(mBase + s * 16 + (lane & 15)) * ISD;
  const v8f zero = {0.f,0.f,0.f,0.f,0.f,0.f,0.f,0.f};
  v8f acc[4][4];
#pragma unroll
  for (int s = 0; s < 4; ++s)
#pragma unroll
    for (int j = 0; j < 4; ++j) acc[s][j] = zero;
  for (int kk = 0; kk < ISD; kk += 32) {
    __builtin_prefetch(aRow[0] + kk + 64, 0, 1);
    v16bf a0 = load_a_frag(aRow[0], kk, lane);
    v16bf a1 = load_a_frag(aRow[1], kk, lane);
    v16bf a2 = load_a_frag(aRow[2], kk, lane);
    v16bf a3 = load_a_frag(aRow[3], kk, lane);
#pragma unroll
    for (int j = 0; j < 4; ++j) {
      v16bf b = load_b_frag(wdT, ISD, nBase + j * 16, kk, lane);
      acc[0][j] = wmma_bf16(a0, b, acc[0][j]);
      acc[1][j] = wmma_bf16(a1, b, acc[1][j]);
      acc[2][j] = wmma_bf16(a2, b, acc[2][j]);
      acc[3][j] = wmma_bf16(a3, b, acc[3][j]);
    }
  }
  const int mOff = (lane >> 4) * 8;
  const int nCol = lane & 15;
#pragma unroll
  for (int s = 0; s < 4; ++s)
#pragma unroll
    for (int j = 0; j < 4; ++j)
#pragma unroll
      for (int r = 0; r < 8; ++r) {
        const int m = mBase + s * 16 + mOff + r;
        const int n = nBase + j * 16 + nCol;
        out[(size_t)m * HD + n] = RATIO * acc[s][j][r];
      }
}

// ---------------- kernel 6: routed gate+up (gathered rows), wave tile 32M x 64N ----------------
// hr[entry][ID] = silu(x_tok @ wg[e]) * (x_tok @ wu[e]); entry = 2*token + slot
__global__ void __launch_bounds__(128) k_gateup_routed(const __bf16* __restrict__ xb,
                                                       const __bf16* __restrict__ wgT,  // [E][ID][HD]
                                                       const __bf16* __restrict__ wuT,
                                                       const int* __restrict__ counts,
                                                       const int* __restrict__ lists,
                                                       __bf16* __restrict__ hr) {
  const int e = blockIdx.z;
  const int cnt = counts[e];
  const int mBase = blockIdx.y * 32;
  if (mBase >= cnt) return;
  const int* lst = lists + e * Ttok;
  const int lane = threadIdx.x & 31, wave = threadIdx.x >> 5;
  const int nBase = blockIdx.x * 256 + wave * 64;
  const __bf16* aRow[2];
#pragma unroll
  for (int s = 0; s < 2; ++s) {
    int ar = mBase + s * 16 + (lane & 15);
    if (ar >= cnt) ar = cnt - 1;
    aRow[s] = xb + (size_t)(lst[ar] >> 1) * HD;
  }
  const __bf16* wg = wgT + (size_t)e * ID * HD;
  const __bf16* wu = wuT + (size_t)e * ID * HD;
  const v8f zero = {0.f,0.f,0.f,0.f,0.f,0.f,0.f,0.f};
  v8f accg[2][4], accu[2][4];
#pragma unroll
  for (int s = 0; s < 2; ++s)
#pragma unroll
    for (int j = 0; j < 4; ++j) { accg[s][j] = zero; accu[s][j] = zero; }
  for (int kk = 0; kk < HD; kk += 32) {
    __builtin_prefetch(aRow[0] + kk + 64, 0, 1);
    v16bf a0 = load_a_frag(aRow[0], kk, lane);
    v16bf a1 = load_a_frag(aRow[1], kk, lane);
#pragma unroll
    for (int j = 0; j < 4; ++j) {
      v16bf bg = load_b_frag(wg, HD, nBase + j * 16, kk, lane);
      accg[0][j] = wmma_bf16(a0, bg, accg[0][j]);
      accg[1][j] = wmma_bf16(a1, bg, accg[1][j]);
      v16bf bu = load_b_frag(wu, HD, nBase + j * 16, kk, lane);
      accu[0][j] = wmma_bf16(a0, bu, accu[0][j]);
      accu[1][j] = wmma_bf16(a1, bu, accu[1][j]);
    }
  }
  const int mOff = (lane >> 4) * 8;
  const int nCol = lane & 15;
#pragma unroll
  for (int s = 0; s < 2; ++s) {
    int ent[8]; bool ok[8];
#pragma unroll
    for (int r = 0; r < 8; ++r) {
      const int rr = mBase + s * 16 + mOff + r;
      ok[r] = (rr < cnt);
      ent[r] = lst[ok[r] ? rr : 0];
    }
#pragma unroll
    for (int j = 0; j < 4; ++j)
#pragma unroll
      for (int r = 0; r < 8; ++r)
        if (ok[r]) {
          const int n = nBase + j * 16 + nCol;
          hr[(size_t)ent[r] * ID + n] = (__bf16)(silu_f(accg[s][j][r]) * accu[s][j][r]);
        }
  }
}

// ---------------- kernel 7: routed down-proj, wave tile 64M x 64N -> yr[entry][HD] * weight ----------------
__global__ void __launch_bounds__(128) k_down_routed(const __bf16* __restrict__ hr,
                                                     const __bf16* __restrict__ wdT,  // [E][HD][ID]
                                                     const int* __restrict__ counts,
                                                     const int* __restrict__ lists,
                                                     const float* __restrict__ wsel,
                                                     float* __restrict__ yr) {
  const int e = blockIdx.z;
  const int cnt = counts[e];
  const int mBase = blockIdx.y * 64;
  if (mBase >= cnt) return;
  const int* lst = lists + e * Ttok;
  const int lane = threadIdx.x & 31, wave = threadIdx.x >> 5;
  const int nBase = blockIdx.x * 256 + wave * 64;
  const __bf16* aRow[4];
#pragma unroll
  for (int s = 0; s < 4; ++s) {
    int ar = mBase + s * 16 + (lane & 15);
    if (ar >= cnt) ar = cnt - 1;
    aRow[s] = hr + (size_t)lst[ar] * ID;
  }
  const __bf16* wd = wdT + (size_t)e * HD * ID;
  const v8f zero = {0.f,0.f,0.f,0.f,0.f,0.f,0.f,0.f};
  v8f acc[4][4];
#pragma unroll
  for (int s = 0; s < 4; ++s)
#pragma unroll
    for (int j = 0; j < 4; ++j) acc[s][j] = zero;
  for (int kk = 0; kk < ID; kk += 32) {
    __builtin_prefetch(aRow[0] + kk + 64, 0, 1);
    v16bf a0 = load_a_frag(aRow[0], kk, lane);
    v16bf a1 = load_a_frag(aRow[1], kk, lane);
    v16bf a2 = load_a_frag(aRow[2], kk, lane);
    v16bf a3 = load_a_frag(aRow[3], kk, lane);
#pragma unroll
    for (int j = 0; j < 4; ++j) {
      v16bf b = load_b_frag(wd, ID, nBase + j * 16, kk, lane);
      acc[0][j] = wmma_bf16(a0, b, acc[0][j]);
      acc[1][j] = wmma_bf16(a1, b, acc[1][j]);
      acc[2][j] = wmma_bf16(a2, b, acc[2][j]);
      acc[3][j] = wmma_bf16(a3, b, acc[3][j]);
    }
  }
  const int mOff = (lane >> 4) * 8;
  const int nCol = lane & 15;
#pragma unroll
  for (int s = 0; s < 4; ++s)
#pragma unroll
    for (int r = 0; r < 8; ++r) {
      const int rr = mBase + s * 16 + mOff + r;
      if (rr < cnt) {
        const int entD = lst[rr];
        const float w = wsel[entD];
#pragma unroll
        for (int j = 0; j < 4; ++j) {
          const int n = nBase + j * 16 + nCol;
          yr[(size_t)entD * HD + n] = w * acc[s][j][r];
        }
      }
    }
}

// ---------------- kernel 8: combine shared + 2 routed slots ----------------
__global__ void k_combine(float* __restrict__ out, const float* __restrict__ yr) {
  const size_t i = (size_t)blockIdx.x * 256 + threadIdx.x;
  const size_t t = i / HD, c = i % HD;
  out[i] += yr[(2 * t) * HD + c] + yr[(2 * t + 1) * HD + c];
}

// ---------------- host launcher ----------------
extern "C" void kernel_launch(void* const* d_in, const int* in_sizes, int n_in,
                              void* d_out, int out_size, void* d_ws, size_t ws_size,
                              hipStream_t stream) {
  (void)in_sizes; (void)n_in; (void)out_size; (void)ws_size;
  const float* x     = (const float*)d_in[0];   // [T][HD]
  const float* cent  = (const float*)d_in[1];   // [NE][HD]
  const float* gbias = (const float*)d_in[2];   // [NE]
  const float* wg_s  = (const float*)d_in[3];   // [HD][ISD]
  const float* wu_s  = (const float*)d_in[4];   // [HD][ISD]
  const float* wd_s  = (const float*)d_in[5];   // [ISD][HD]
  const float* wg    = (const float*)d_in[6];   // [NE][HD][ID]
  const float* wu    = (const float*)d_in[7];   // [NE][HD][ID]
  const float* wd    = (const float*)d_in[8];   // [NE][ID][HD]
  float* out = (float*)d_out;                   // [T][HD]

  // -------- workspace layout (bytes), 256B aligned --------
  char* ws = (char*)d_ws;
  size_t off = 0;
  auto take = [&](size_t bytes) { char* p = ws + off; off = (off + bytes + 255) & ~(size_t)255; return p; };
  __bf16* xb   = (__bf16*)take((size_t)Ttok * HD * 2);          // 16 MB
  __bf16* wgsT = (__bf16*)take((size_t)ISD * HD * 2);           //  8 MB
  __bf16* wusT = (__bf16*)take((size_t)ISD * HD * 2);           //  8 MB
  __bf16* wdsT = (__bf16*)take((size_t)HD * ISD * 2);           //  8 MB
  __bf16* wgT  = (__bf16*)take((size_t)NE * ID * HD * 2);       // 64 MB
  __bf16* wuT  = (__bf16*)take((size_t)NE * ID * HD * 2);       // 64 MB
  __bf16* wdT  = (__bf16*)take((size_t)NE * HD * ID * 2);       // 64 MB
  __bf16* hs   = (__bf16*)take((size_t)Ttok * ISD * 2);         // 16 MB
  __bf16* hr   = (__bf16*)take((size_t)Ttok * TOPK * ID * 2);   // 16 MB
  float*  yr   = (float*) take((size_t)Ttok * TOPK * HD * 4);   // 64 MB
  float*  wsel = (float*) take((size_t)Ttok * TOPK * 4);
  int*    cnts = (int*)   take(256);
  int*    lsts = (int*)   take((size_t)NE * Ttok * 4);

  // 0) counters
  k_zero_counts<<<1, 32, 0, stream>>>(cnts);
  // 1) activations -> bf16
  k_cvt_x<<<(Ttok * HD) / 256, 256, 0, stream>>>(x, xb, Ttok * HD);
  // 2) weights -> transposed bf16
  dim3 tb(32, 8);
  k_transpose_cvt<<<dim3(ISD / 32, HD / 32, 1),  tb, 0, stream>>>(wg_s, wgsT, HD,  ISD);
  k_transpose_cvt<<<dim3(ISD / 32, HD / 32, 1),  tb, 0, stream>>>(wu_s, wusT, HD,  ISD);
  k_transpose_cvt<<<dim3(HD / 32,  ISD / 32, 1), tb, 0, stream>>>(wd_s, wdsT, ISD, HD);
  k_transpose_cvt<<<dim3(ID / 32,  HD / 32, NE), tb, 0, stream>>>(wg,   wgT,  HD,  ID);
  k_transpose_cvt<<<dim3(ID / 32,  HD / 32, NE), tb, 0, stream>>>(wu,   wuT,  HD,  ID);
  k_transpose_cvt<<<dim3(HD / 32,  ID / 32, NE), tb, 0, stream>>>(wd,   wdT,  ID,  HD);
  // 3) routing: top-2 selection + gather lists + clean weights
  k_routing<<<Ttok, 128, 0, stream>>>(x, cent, gbias, wsel, cnts, lsts);
  // 4) shared expert SwiGLU (WMMA, 32x64 tiles) -> hs
  k_gateup_shared<<<dim3(ISD / 256, Ttok / 32), 128, 0, stream>>>(xb, wgsT, wusT, hs);
  // 5) shared down-proj (WMMA, 64x64 tiles) -> out = RATIO * (hs @ wd_s)
  k_down_shared<<<dim3(HD / 256, Ttok / 64), 128, 0, stream>>>(hs, wdsT, out);
  // 6) routed gate+up (WMMA, gathered, 32x64 tiles) -> hr
  k_gateup_routed<<<dim3(ID / 256, Ttok / 32, NE), 128, 0, stream>>>(xb, wgT, wuT, cnts, lsts, hr);
  // 7) routed down-proj (WMMA, gathered, 64x64 tiles) -> yr (routing weight applied)
  k_down_routed<<<dim3(HD / 256, Ttok / 64, NE), 128, 0, stream>>>(hr, wdT, cnts, lsts, wsel, yr);
  // 8) combine
  k_combine<<<(Ttok * HD) / 256, 256, 0, stream>>>(out, yr);
}
